// LSTM_intensity_53051436040335
// MI455X (gfx1250) — compile-verified, matrix operand order checked by reading, not run on hardware
//
#include <hip/hip_runtime.h>

// ---------------------------------------------------------------------------
// LSTM (B=256, IN=256, HID=512, T=512) persistent-kernel for gfx1250:
// bf16 WMMA (v_wmma_f32_16x16x32_bf16) for input + recurrent GEMMs.
// - Weights for each workgroup's hidden tile staged once into LDS (96 KB of
//   the 320 KB WGP LDS); per-step B fragments are ds_load_b128.
// - h double-buffered in L2-resident bf16 buffers -> ONE grid barrier/step.
// - c-state in registers; tiny dense head done by 256 waves with shfl_xor.
// ---------------------------------------------------------------------------

#define T_STEPS 512
#define BATCH   256
#define INSZ    256
#define HID     512
#define NWG     64      // persistent workgroups (64 x 8 waves = 512 waves = #tiles)

typedef __attribute__((ext_vector_type(16))) __bf16 v16bf;
typedef __attribute__((ext_vector_type(8)))  __bf16 v8bf;
typedef __attribute__((ext_vector_type(8)))  float  v8f;

union Frag {
  v16bf v;
  v8bf  h[2];
};

__device__ __forceinline__ float sigmoid_f(float x) {
  return 1.0f / (1.0f + __expf(-x));
}
__device__ __forceinline__ float tanh_f(float x) {
  return 1.0f - 2.0f / (__expf(2.0f * x) + 1.0f);   // saturates for |x| large
}

__device__ __forceinline__ void grid_barrier(unsigned* cnt, unsigned* round) {
  __syncthreads();
  if (threadIdx.x == 0) {
    __threadfence();
    unsigned target = (++(*round)) * NWG;
    atomicAdd(cnt, 1u);
    while (__hip_atomic_load(cnt, __ATOMIC_RELAXED, __HIP_MEMORY_SCOPE_AGENT) < target) {
      __builtin_amdgcn_s_sleep(1);
    }
    __threadfence();
  }
  __syncthreads();
}

// --------------------------- repack kernels --------------------------------

__global__ void k_f32_to_bf16(__bf16* __restrict__ dst, const float* __restrict__ src, int n) {
  int i = blockIdx.x * blockDim.x + threadIdx.x;
  if (i < n) dst[i] = (__bf16)src[i];
}

__global__ void k_zero_u32(unsigned* p) {
  if (threadIdx.x == 0 && blockIdx.x == 0) *p = 0u;
}

// --------------------------- main persistent kernel ------------------------

__global__ __launch_bounds__(256) void lstm_persistent(
    const __bf16* __restrict__ xb,    // [T, BATCH, INSZ] bf16
    const __bf16* __restrict__ wih,   // [4*HID, INSZ]    bf16
    const __bf16* __restrict__ whh,   // [4*HID, HID]     bf16
    __bf16*       __restrict__ hbuf,  // [2, BATCH, HID]  bf16 (h_t in buf[t&1])
    unsigned*     __restrict__ barcnt,
    const float* __restrict__ b_ih, const float* __restrict__ b_hh,
    const float* __restrict__ c0,
    const float* __restrict__ W1, const float* __restrict__ b1,
    const float* __restrict__ W2, const float* __restrict__ b2,
    float* __restrict__ out)          // [T, BATCH]
{
  // LDS-staged weights for this workgroup's hidden tile j:
  // 64 gate-rows (4 gates x 16 columns), K-major, row-major.
  __shared__ __bf16 sWih[64 * INSZ];  // 32 KB
  __shared__ __bf16 sWhh[64 * HID];   // 64 KB

  const int tid  = threadIdx.x;
  const int lane = tid & 31;
  const int warp = tid >> 5;
  const int wgid = blockIdx.x * 8 + warp;       // 0..511 (unique wave id)
  const int j    = blockIdx.x >> 1;             // hidden tile 0..31 (shared by WG)
  const int m    = (blockIdx.x & 1) * 8 + warp; // batch tile  0..15
  const int mrow = lane & 15;
  const int half = lane >> 4;
  const int nj   = j * 16 + mrow;               // owned hidden column (C-layout)
  const int arow = m * 16 + mrow;               // batch row for A-layout

  // ---- one-time LDS fill: gate rows g*HID + j*16 + r  ->  local row g*16+r ----
  {
    // W_hh: 64 rows x 512 bf16 = 64 x 64 16B-chunks
    for (int cidx = tid; cidx < 64 * 64; cidx += 256) {
      int row = cidx >> 6, c = cidx & 63;
      int g = row >> 4, r = row & 15;
      int grow = g * HID + j * 16 + r;
      ((uint4*)sWhh)[cidx] = ((const uint4*)(whh + (size_t)grow * HID))[c];
    }
    // W_ih: 64 rows x 256 bf16 = 64 x 32 16B-chunks
    for (int cidx = tid; cidx < 64 * 32; cidx += 256) {
      int row = cidx >> 5, c = cidx & 31;
      int g = row >> 4, r = row & 15;
      int grow = g * HID + j * 16 + r;
      ((uint4*)sWih)[cidx] = ((const uint4*)(wih + (size_t)grow * INSZ))[c];
    }
    __syncthreads();
  }

  // per-gate bias for this lane's output column (constant over time)
  float bias[4];
#pragma unroll
  for (int g = 0; g < 4; ++g) bias[g] = b_ih[g * HID + nj] + b_hh[g * HID + nj];

  // cell state in registers, C/D VGPR layout: (row = m*16 + r + half*8, col = nj)
  float c[8];
#pragma unroll
  for (int r = 0; r < 8; ++r) {
    int row = m * 16 + r + half * 8;
    c[r] = c0[row * HID + nj];
  }

  // per-lane B-fragment row pointers into LDS (lane's N column = mrow)
  const __bf16* wihRow[4];
  const __bf16* whhRow[4];
#pragma unroll
  for (int g = 0; g < 4; ++g) {
    wihRow[g] = sWih + (g * 16 + mrow) * INSZ;
    whhRow[g] = sWhh + (g * 16 + mrow) * HID;
  }

  unsigned round = 0;

  for (int t = 0; t < T_STEPS; ++t) {
    // double-buffered h: read h_t from buf[t&1], write h_{t+1} to buf[(t+1)&1].
    // The buffer being overwritten holds h_{t-1}, whose last readers finished
    // before barrier (t-1) => a single barrier per step suffices.
    const __bf16* hcur = hbuf + (size_t)(t & 1) * BATCH * HID;
    __bf16*       hnxt = hbuf + (size_t)((t + 1) & 1) * BATCH * HID;

    v8f acc[4] = {};   // f32 accumulators, one per gate

    // ---- phase 1: gates += x_t @ W_ih^T   (K = 256), B from LDS ----
    const __bf16* xrow = xb + (size_t)t * BATCH * INSZ + (size_t)arow * INSZ;
#pragma unroll 2
    for (int kb = 0; kb < INSZ; kb += 32) {
      Frag a;
      a.h[0] = *(const v8bf*)(xrow + kb + half * 8);
      a.h[1] = *(const v8bf*)(xrow + kb + 16 + half * 8);
#pragma unroll
      for (int g = 0; g < 4; ++g) {
        Frag b;
        b.h[0] = *(const v8bf*)(wihRow[g] + kb + half * 8);
        b.h[1] = *(const v8bf*)(wihRow[g] + kb + 16 + half * 8);
        acc[g] = __builtin_amdgcn_wmma_f32_16x16x32_bf16(
            false, a.v, false, b.v, (short)0, acc[g], false, false);
      }
    }

    // ---- phase 2: gates += h_t @ W_hh^T   (K = 512), B from LDS ----
    const __bf16* hrow = hcur + (size_t)arow * HID;
#pragma unroll 2
    for (int kb = 0; kb < HID; kb += 32) {
      Frag a;
      a.h[0] = *(const v8bf*)(hrow + kb + half * 8);
      a.h[1] = *(const v8bf*)(hrow + kb + 16 + half * 8);
#pragma unroll
      for (int g = 0; g < 4; ++g) {
        Frag b;
        b.h[0] = *(const v8bf*)(whhRow[g] + kb + half * 8);
        b.h[1] = *(const v8bf*)(whhRow[g] + kb + 16 + half * 8);
        acc[g] = __builtin_amdgcn_wmma_f32_16x16x32_bf16(
            false, a.v, false, b.v, (short)0, acc[g], false, false);
      }
    }

    // ---- cell update (pure register math) + publish h_{t+1} ----
#pragma unroll
    for (int r = 0; r < 8; ++r) {
      float iv = sigmoid_f(acc[0][r] + bias[0]);
      float fv = sigmoid_f(acc[1][r] + bias[1]);
      float gv = tanh_f(acc[2][r] + bias[2]);
      float ov = sigmoid_f(acc[3][r] + bias[3]);
      float cv = fv * c[r] + iv * gv;
      c[r]     = cv;
      float hv = ov * tanh_f(cv);
      int row  = m * 16 + r + half * 8;
      hnxt[row * HID + nj] = (__bf16)hv;
    }

    // single barrier per step: h_{t+1} fully published
    grid_barrier(barcnt, &round);

    // ---- dense head: y_t[b] = elu((h_{t+1}[b] @ W1^T + b1) @ W2^T + b2) ----
    // waves 0..255 each own one batch row; wave-uniform branch (no WMMA inside)
    if (wgid < BATCH) {
      const int b = wgid;
      const __bf16* hr = hnxt + (size_t)b * HID + lane * 16;
      Frag hv;
      hv.h[0] = *(const v8bf*)(hr);
      hv.h[1] = *(const v8bf*)(hr + 8);
      float y5[5];
#pragma unroll
      for (int u = 0; u < 5; ++u) {
        const float* w1r = W1 + u * HID + lane * 16;
        float s = 0.0f;
#pragma unroll
        for (int k = 0; k < 16; ++k) s += (float)hv.v[k] * w1r[k];
        y5[u] = s;
      }
#pragma unroll
      for (int u = 0; u < 5; ++u) {
#pragma unroll
        for (int off = 16; off >= 1; off >>= 1)
          y5[u] += __shfl_xor(y5[u], off, 32);
      }
      if (lane == 0) {
        float z = b2[0];
#pragma unroll
        for (int u = 0; u < 5; ++u) z += (y5[u] + b1[u]) * W2[u];
        out[t * BATCH + b] = (z > 0.0f) ? z : (__expf(z) - 1.0f);
      }
    }
  }
}

// --------------------------- launch ----------------------------------------

extern "C" void kernel_launch(void* const* d_in, const int* in_sizes, int n_in,
                              void* d_out, int out_size, void* d_ws, size_t ws_size,
                              hipStream_t stream) {
  (void)in_sizes; (void)n_in; (void)out_size; (void)ws_size;

  const float* x    = (const float*)d_in[0];   // [512,256,256]
  const float* h0   = (const float*)d_in[1];   // [256,512]
  const float* c0   = (const float*)d_in[2];   // [256,512]
  const float* W_ih = (const float*)d_in[3];   // [2048,256]
  const float* W_hh = (const float*)d_in[4];   // [2048,512]
  const float* b_ih = (const float*)d_in[5];   // [2048]
  const float* b_hh = (const float*)d_in[6];   // [2048]
  const float* W1   = (const float*)d_in[7];   // [5,512]
  const float* b1   = (const float*)d_in[8];   // [5]
  const float* W2   = (const float*)d_in[9];   // [1,5]
  const float* b2   = (const float*)d_in[10];  // [1]

  // workspace layout (16B-aligned chunks)
  char* ws = (char*)d_ws;
  const size_t N_H   = (size_t)BATCH * HID;            // 131072
  const size_t N_WIH = (size_t)4 * HID * INSZ;         // 524288
  const size_t N_WHH = (size_t)4 * HID * HID;          // 1048576
  const size_t N_X   = (size_t)T_STEPS * BATCH * INSZ; // 33554432

  unsigned* barcnt = (unsigned*)ws;                    // 256 B reserved
  __bf16* hbuf = (__bf16*)(ws + 256);                  // [2, BATCH, HID]
  __bf16* wih  = (__bf16*)((char*)hbuf + 2 * N_H * 2);
  __bf16* whh  = (__bf16*)((char*)wih  + N_WIH * 2);
  __bf16* xb   = (__bf16*)((char*)whh  + N_WHH * 2);

  // one-time (per launch) repack: f32 -> bf16, zero barrier counter.
  // h_0 goes into buffer 0 (t=0 reads buf[0]).
  k_zero_u32<<<1, 32, 0, stream>>>(barcnt);
  k_f32_to_bf16<<<(int)((N_H   + 255) / 256), 256, 0, stream>>>(hbuf, h0,   (int)N_H);
  k_f32_to_bf16<<<(int)((N_WIH + 255) / 256), 256, 0, stream>>>(wih,  W_ih, (int)N_WIH);
  k_f32_to_bf16<<<(int)((N_WHH + 255) / 256), 256, 0, stream>>>(whh,  W_hh, (int)N_WHH);
  k_f32_to_bf16<<<(int)((N_X   + 255) / 256), 256, 0, stream>>>(xb,   x,    (int)N_X);

  lstm_persistent<<<NWG, 256, 0, stream>>>(
      xb, wih, whh, hbuf, barcnt,
      b_ih, b_hh, c0, W1, b1, W2, b2,
      (float*)d_out);
}
